// TemporalKernelAttentionKAN_48215302865005
// MI455X (gfx1250) — matrix operand order repaired and checked by reading, not compiled, not run on hardware
//
#include <hip/hip_runtime.h>
#include <hip/hip_bf16.h>

// ---------------------------------------------------------------------------
// Problem constants (match reference)
// ---------------------------------------------------------------------------
#define BATCH   32768
#define LLAG    512
#define DATT    256
#define COEF    11          // spline basis functions per input feature
#define KPF     12          // 1 (silu/base) + 11 (spline) K-slots per feature
#define KTOT    (DATT * KPF)   // 3072

// Fused-layer tiling
#define BM      128         // batch rows per block
#define FCH     8           // input features per K-chunk
#define KCH     (FCH * KPF) // 96 K per chunk  -> 3 WMMA K-steps of 32
#define NCHUNKS (DATT / FCH)// 32
#define ASTR    104         // A LDS row stride (ushorts): 96 + pad, 16B-mult
#define BSTR    264         // B LDS row stride (ushorts): 256 + 4dw pad (TDM)

// LDS pool layout (manual so TDM descriptor lds_addr is a constant offset)
#define A_OFF   0
#define A_BYTES (BM * ASTR * 2)          // 26624
#define B0_OFF  A_BYTES                  // 26624
#define B_BYTES (KCH * BSTR * 2)         // 50688
#define B1_OFF  (B0_OFF + B_BYTES)       // 77312
#define POOL_BYTES (B1_OFF + B_BYTES)    // 128000

typedef __attribute__((ext_vector_type(16))) __bf16 v16bfx;
typedef __attribute__((ext_vector_type(8)))  float  v8fx;
typedef unsigned int u32x4 __attribute__((ext_vector_type(4)));
typedef int          i32x8 __attribute__((ext_vector_type(8)));
typedef int          i32x4 __attribute__((ext_vector_type(4)));

// Tensor Data Mover builtin: arity differs between toolchains (bridge doc).
#if __has_include(<hip/amd_detail/amd_gfx1250_TDM.h>)
#define TDM_LOAD(g0, g1, g2, g3) \
    __builtin_amdgcn_tensor_load_to_lds((g0), (g1), (g2), (g3), \
        (i32x8){0, 0, 0, 0, 0, 0, 0, 0}, 0)
#else
#define TDM_LOAD(g0, g1, g2, g3) \
    __builtin_amdgcn_tensor_load_to_lds((g0), (g1), (g2), (g3), 0)
#endif

// ---------------------------------------------------------------------------
// Helpers
// ---------------------------------------------------------------------------
__device__ __forceinline__ unsigned short f2bf(float f) {
    union { float f; unsigned u; } v; v.f = f;
    unsigned r = v.u + 0x7FFFu + ((v.u >> 16) & 1u);   // round-to-nearest-even
    return (unsigned short)(r >> 16);
}

__device__ __forceinline__ float silu(float x) {
    return x / (1.0f + __expf(-x));
}

__device__ __forceinline__ float softplus(float x) {
    return (x > 20.0f) ? x : __logf(1.0f + __expf(x));
}

// Cox–de Boor recursion: order 3, grid_size 8, grid_range (-1,1), h = 0.25.
__device__ __forceinline__ void kan_bases(float x, float* bo) {
    const float h = 0.25f;
    float t[15];
#pragma unroll
    for (int j = 0; j < 15; ++j) t[j] = -1.0f + (float)(j - 3) * h;
    float b[14];
#pragma unroll
    for (int j = 0; j < 14; ++j)
        b[j] = (x >= t[j] && x < t[j + 1]) ? 1.0f : 0.0f;
#pragma unroll
    for (int k = 1; k <= 3; ++k) {
        float inv = 1.0f / ((float)k * h);
#pragma unroll
        for (int j = 0; j < 14 - k; ++j)
            b[j] = ((x - t[j]) * b[j] + (t[j + k + 1] - x) * b[j + 1]) * inv;
    }
#pragma unroll
    for (int j = 0; j < COEF; ++j) bo[j] = b[j];
}

// ---------------------------------------------------------------------------
// Kernel 1: temporal kernel weights + normalization + weighted sum.
// ---------------------------------------------------------------------------
__global__ __launch_bounds__(256) void temporal_weights_kernel(
    const float* __restrict__ y, const float* __restrict__ dt,
    const float* __restrict__ pil, const float* __restrict__ lamr,
    float* __restrict__ wl_out, float* __restrict__ s_out)
{
    __shared__ float red1[256];
    __shared__ float red2[256];
    const int b   = blockIdx.x;
    const int tid = threadIdx.x;

    float p0 = pil[0], p1 = pil[1], p2 = pil[2], p3 = pil[3];
    float mx = fmaxf(fmaxf(p0, p1), fmaxf(p2, p3));
    float e0 = __expf(p0 - mx), e1 = __expf(p1 - mx);
    float e2 = __expf(p2 - mx), e3 = __expf(p3 - mx);
    float inv_se = 1.0f / (e0 + e1 + e2 + e3);
    float pi0 = e0 * inv_se, pi1 = e1 * inv_se, pi2 = e2 * inv_se, pi3 = e3 * inv_se;
    float l0 = softplus(lamr[0]) + 1e-6f, l1 = softplus(lamr[1]) + 1e-6f;
    float l2 = softplus(lamr[2]) + 1e-6f, l3 = softplus(lamr[3]) + 1e-6f;

    const size_t base = (size_t)b * LLAG;
    float d0 = dt[base + tid], d1 = dt[base + tid + 256];
    float y0 = y[base + tid],  y1 = y[base + tid + 256];

    float w0 = pi0 * __expf(-l0 * d0) + pi1 * __expf(-l1 * d0)
             + pi2 * __expf(-l2 * d0) + pi3 * __expf(-l3 * d0);
    float w1 = pi0 * __expf(-l0 * d1) + pi1 * __expf(-l1 * d1)
             + pi2 * __expf(-l2 * d1) + pi3 * __expf(-l3 * d1);

    red1[tid] = w0 + w1;
    red2[tid] = w0 * y0 + w1 * y1;
    __syncthreads();
#pragma unroll
    for (int s = 128; s > 0; s >>= 1) {
        if (tid < s) { red1[tid] += red1[tid + s]; red2[tid] += red2[tid + s]; }
        __syncthreads();
    }
    float inv = 1.0f / (red1[0] + 1e-8f);
    wl_out[base + tid]       = w0 * inv;
    wl_out[base + tid + 256] = w1 * inv;
    if (tid == 0) s_out[b] = red2[0] * inv;
}

// ---------------------------------------------------------------------------
// Kernel 2: prepack one 256->256 KAN layer's weights into K-major bf16
// Wpk[3072][256]; row k = i*12 + c (c==0 -> base weight, else spline*scaler).
// ---------------------------------------------------------------------------
__global__ __launch_bounds__(256) void prepack_kernel(
    const float* __restrict__ bw, const float* __restrict__ sw,
    const float* __restrict__ sc, unsigned short* __restrict__ wpk)
{
    int idx = blockIdx.x * 256 + threadIdx.x;
    int k = idx >> 8;
    int o = idx & 255;
    int i = k / KPF, c = k % KPF;
    float v = (c == 0) ? bw[o * DATT + i]
                       : sw[(o * DATT + i) * COEF + (c - 1)] * sc[o * DATT + i];
    wpk[idx] = f2bf(v);
}

// ---------------------------------------------------------------------------
// Kernel 3: KAN layer 0 (in=1 -> out=256).
// ---------------------------------------------------------------------------
__global__ __launch_bounds__(256) void kan_layer0_kernel(
    const float* __restrict__ s, const float* __restrict__ bw,
    const float* __restrict__ sw, const float* __restrict__ sc,
    float* __restrict__ z)
{
    int b = blockIdx.x;
    int o = threadIdx.x;
    float x = s[b];
    float bas[COEF];
    kan_bases(x, bas);
    float acc = silu(x) * bw[o];
    float scale = sc[o];
#pragma unroll
    for (int c = 0; c < COEF; ++c) acc += bas[c] * sw[o * COEF + c] * scale;
    z[(size_t)b * DATT + o] = acc;
}

// ---------------------------------------------------------------------------
// Kernel 4: fused 256->256 KAN layer: bf16 WMMA + TDM-streamed weights.
//   out[b,o] = sum_k A[b,k] * W[k,o],  K = 3072 = 256 feats x (silu + 11
//   bases). Basis computation is fused into LDS (A never hits HBM).
//   Weight slabs (96 x 256 bf16 + hardware LDS padding) are DMA'd by the
//   Tensor Data Mover, double-buffered against compute, synchronized with
//   s_wait_tensorcnt + workgroup barrier. The 48 WMMAs per chunk run as one
//   flat software pipeline with a 4-deep B-fragment rotation so LDS latency
//   hides behind ~3 matrix ops instead of stalling dscnt==0 per WMMA.
// ---------------------------------------------------------------------------
__global__ __launch_bounds__(256) void kan_fused_layer_kernel(
    const float* __restrict__ xin,              // [B, 256]
    const unsigned short* __restrict__ wpk,     // [3072, 256] bf16, K-major
    float* __restrict__ out)                    // [B, 256]
{
    __shared__ __align__(16) unsigned char POOL[POOL_BYTES];
    unsigned short* const Albs = (unsigned short*)(POOL + A_OFF);
    unsigned short* const Bb0  = (unsigned short*)(POOL + B0_OFF);
    unsigned short* const Bb1  = (unsigned short*)(POOL + B1_OFF);

    const int tid  = threadIdx.x;
    const int wave = tid >> 5;
    const int lane = tid & 31;
    const int m    = lane & 15;
    const int hi   = lane >> 4;
    const int row0 = blockIdx.x * BM;

    // TDM D# group1 (constant part): data_size=2B (code 1), pad_enable,
    // pad_interval code 6 (pad every 128 DWORDs = one 256-bf16 row),
    // pad_amount code 3 (4 DWORDs = 16B) -> LDS row stride 264 ushorts.
    const int g1w0 = (1 << 16) | (1 << 20) | (6 << 22) | (3 << 25);
    const i32x8 g1 = { g1w0,
                       (int)((DATT & 0xFFFF) << 16),          // tensor_dim0 lo
                       (int)(((unsigned)KTOT & 0xFFFF) << 16),// dim0 hi | dim1 lo
                       (int)((DATT & 0xFFFF) << 16),          // dim1 hi | tile_dim0
                       (int)KCH,                              // tile_dim1 | tile_dim2=0
                       (int)DATT,                             // dim0_stride lo
                       0, 0 };
    const i32x4 gz = { 0, 0, 0, 0 };

    v8fx acc[16] = {};

    // Prologue: DMA weight chunk 0 into buffer 0 (one wave issues; EXEC is
    // ignored by TENSOR ops, so gate with a wave-level branch).
    if (tid < 32) {
        unsigned long long ga = (unsigned long long)(size_t)wpk;
        u32x4 g0 = { 1u,                       // count=1, no gather
                     (unsigned)B0_OFF,         // lds_addr
                     (unsigned)(ga & 0xFFFFFFFFu),
                     (unsigned)((ga >> 32) & 0x1FFFFFFu) | 0x80000000u }; // type=2
        TDM_LOAD(g0, g1, gz, gz);
    }

    for (int ch = 0; ch < NCHUNKS; ++ch) {
        if (ch) __syncthreads();                // prev WMMA LDS reads retired
        const int f0 = ch * FCH;

        // --- activation slab: silu + 11 spline bases -> bf16 in LDS -------
#pragma unroll
        for (int p = 0; p < 4; ++p) {
            int idx  = tid + p * 256;
            int feat = idx & 7;
            int rl   = idx >> 3;
            float x  = xin[(size_t)(row0 + rl) * DATT + f0 + feat];
            float v[KPF];
            v[0] = silu(x);
            kan_bases(x, v + 1);
            unsigned* dst = (unsigned*)&Albs[rl * ASTR + feat * KPF];
#pragma unroll
            for (int i = 0; i < 6; ++i)
                dst[i] = (unsigned)f2bf(v[2 * i]) |
                         ((unsigned)f2bf(v[2 * i + 1]) << 16);
        }

        // --- kick TDM for next chunk into the other buffer, then wait for
        //     this chunk's DMA (in-order per wave: <=1 outstanding left) ----
        if (tid < 32) {
            if (ch + 1 < NCHUNKS) {
                unsigned long long ga = (unsigned long long)(size_t)
                    (wpk + (size_t)(ch + 1) * KCH * DATT);
                u32x4 g0 = { 1u,
                             (unsigned)(((ch + 1) & 1) ? B1_OFF : B0_OFF),
                             (unsigned)(ga & 0xFFFFFFFFu),
                             (unsigned)((ga >> 32) & 0x1FFFFFFu) | 0x80000000u };
                TDM_LOAD(g0, g1, gz, gz);
                __builtin_amdgcn_s_wait_tensorcnt(1);
            } else {
                __builtin_amdgcn_s_wait_tensorcnt(0);
            }
        }
        __syncthreads();

        // --- flat 48-WMMA pipeline: 3 K-steps x 16 N-tiles ----------------
        const unsigned short* const Bcur = (ch & 1) ? Bb1 : Bb0;

        // Preload all three A fragments (lane=M; elems 0-7 -> K+{hi*8},
        // elems 8-15 -> K+16+{hi*8}).
        union Af { v16bfx v; uint4 q[2]; } a[3];
        const unsigned short* brk[3];
#pragma unroll
        for (int ks = 0; ks < 3; ++ks) {
            const unsigned short* ar =
                &Albs[(wave * 16 + m) * ASTR + ks * 32 + hi * 8];
            a[ks].q[0] = *(const uint4*)(ar);
            a[ks].q[1] = *(const uint4*)(ar + 16);
            brk[ks] = &Bcur[(ks * 32 + hi * 8) * BSTR + m]; // B: lane -> N col
        }

        // 4-deep B-fragment rotation: fragment for step t reloads slot t&3
        // right after wmma(t) and is next consumed at wmma(t+4).
        union Bf { v16bfx v; uint4 q[2]; unsigned short u[16]; } fr[4];
#pragma unroll
        for (int i = 0; i < 4; ++i) {            // prime (t=0..3, all ks=0)
#pragma unroll
            for (int j = 0; j < 16; ++j)
                fr[i].u[j] = brk[0][i * 16 + (j + (j >> 3) * 8) * BSTR];
        }
#pragma unroll
        for (int t = 0; t < 48; ++t) {
            const int ks = t >> 4;
            const int nt = t & 15;
            acc[nt] = __builtin_amdgcn_wmma_f32_16x16x32_bf16(
                false, a[ks].v, false, fr[t & 3].v, (short)0, acc[nt],
                false, false);
            if (t + 4 < 48) {
                const int t2  = t + 4;
                const unsigned short* br2 = brk[t2 >> 4] + (t2 & 15) * 16;
#pragma unroll
                for (int j = 0; j < 16; ++j)
                    fr[t & 3].u[j] = br2[(j + (j >> 3) * 8) * BSTR];
            }
        }
    }

    // --- epilogue: C layout VGPR r -> M = hi*8 + r, N = nt*16 + m ---------
    const int rbase = row0 + wave * 16 + hi * 8;
#pragma unroll
    for (int nt = 0; nt < 16; ++nt) {
        int col = nt * 16 + m;
#pragma unroll
        for (int r = 0; r < 8; ++r)
            out[(size_t)(rbase + r) * DATT + col] = acc[nt][r];
    }
}

// ---------------------------------------------------------------------------
// Kernel 5: out head (in=256 -> out=1).
// ---------------------------------------------------------------------------
__global__ __launch_bounds__(256) void kan_out_head_kernel(
    const float* __restrict__ h, const float* __restrict__ bw,
    const float* __restrict__ sw, const float* __restrict__ sc,
    float* __restrict__ e)
{
    __shared__ float red[256];
    int b = blockIdx.x;
    int i = threadIdx.x;
    float x = h[(size_t)b * DATT + i];
    float bas[COEF];
    kan_bases(x, bas);
    float acc = silu(x) * bw[i];
    float scale = sc[i];
#pragma unroll
    for (int c = 0; c < COEF; ++c) acc += bas[c] * sw[i * COEF + c] * scale;
    red[i] = acc;
    __syncthreads();
#pragma unroll
    for (int s = 128; s > 0; s >>= 1) {
        if (i < s) red[i] += red[i + s];
        __syncthreads();
    }
    if (i == 0) e[b] = red[0];
}

// ---------------------------------------------------------------------------
// Launch
// ---------------------------------------------------------------------------
extern "C" void kernel_launch(void* const* d_in, const int* in_sizes, int n_in,
                              void* d_out, int out_size, void* d_ws, size_t ws_size,
                              hipStream_t stream) {
    const float* y_lags    = (const float*)d_in[0];
    const float* dt_lags   = (const float*)d_in[1];
    const float* pi_logits = (const float*)d_in[2];
    const float* lam_raw   = (const float*)d_in[3];
    const float* bw0 = (const float*)d_in[4];
    const float* sw0 = (const float*)d_in[5];
    const float* sc0 = (const float*)d_in[6];
    const float* bw1 = (const float*)d_in[7];
    const float* sw1 = (const float*)d_in[8];
    const float* sc1 = (const float*)d_in[9];
    const float* bw2 = (const float*)d_in[10];
    const float* sw2 = (const float*)d_in[11];
    const float* sc2 = (const float*)d_in[12];
    const float* bw3 = (const float*)d_in[13];
    const float* sw3 = (const float*)d_in[14];
    const float* sc3 = (const float*)d_in[15];

    // Output tuple layout: e_att [B], w_lags [B,512], z_att [B,256]
    float* e_att = (float*)d_out;
    float* wlags = e_att + BATCH;
    float* z_att = wlags + (size_t)BATCH * LLAG;

    // Workspace: s [B] | zbuf [B,256] (reused for h) | wpk1 | wpk2
    float* s_buf = (float*)d_ws;
    float* zbuf  = s_buf + BATCH;
    unsigned short* wpk1 = (unsigned short*)(zbuf + (size_t)BATCH * DATT);
    unsigned short* wpk2 = wpk1 + (size_t)KTOT * DATT;

    prepack_kernel<<<(KTOT * DATT) / 256, 256, 0, stream>>>(bw1, sw1, sc1, wpk1);
    prepack_kernel<<<(KTOT * DATT) / 256, 256, 0, stream>>>(bw2, sw2, sc2, wpk2);

    temporal_weights_kernel<<<BATCH, 256, 0, stream>>>(
        y_lags, dt_lags, pi_logits, lam_raw, wlags, s_buf);

    kan_layer0_kernel<<<BATCH, 256, 0, stream>>>(s_buf, bw0, sw0, sc0, zbuf);

    kan_fused_layer_kernel<<<BATCH / BM, 256, 0, stream>>>(zbuf, wpk1, z_att);
    kan_fused_layer_kernel<<<BATCH / BM, 256, 0, stream>>>(z_att, wpk2, zbuf);

    kan_out_head_kernel<<<BATCH, 256, 0, stream>>>(zbuf, bw3, sw3, sc3, e_att);
}
